// JointCascadeGlobalPointer_73315091742845
// MI455X (gfx1250) — matrix operand order repaired and checked by reading, not compiled, not run on hardware
//
#include <hip/hip_runtime.h>
#include <math.h>

// ---------------------------------------------------------------------------
// JointCascadeGlobalPointer for MI455X (gfx1250, wave32, WMMA bf16)
// Shapes: B=4 S=1024 H=768 NH=4 HD=192 E=9 R=12 D=64
// All GEMMs use v_wmma_f32_16x16x32_bf16 (fp32 in/out, bf16 operands, f32 acc)
// global_prefetch_b8 used to hide latency of the next K-step's tiles.
// ---------------------------------------------------------------------------

typedef __attribute__((ext_vector_type(16))) __bf16 v16bf;
typedef __attribute__((ext_vector_type(8)))  float  v8f;

#define H_DIM   768
#define NHEADS  4
#define HD_DIM  192
#define E_T     9
#define R_T     12
#define D_ROT   64
#define BATCH   4
#define SEQ     1024
#define ROWS    (BATCH * SEQ)       // 4096

static __device__ __forceinline__ v8f v8zero() {
  v8f z;
#pragma unroll
  for (int i = 0; i < 8; ++i) z[i] = 0.0f;
  return z;
}

static __device__ __forceinline__ v8f wmma_bf16(v16bf a, v16bf b, v8f c) {
  // emits v_wmma_f32_16x16x32_bf16
  return __builtin_amdgcn_wmma_f32_16x16x32_bf16(false, a, false, b, (short)0, c,
                                                 false, false);
}

// A-fragment (16x32, bf16) from a row-major fp32 row pointer.
// lane M = lane&15 (caller bakes into rowPtr), element j -> k = k0 + half*16 + j
static __device__ __forceinline__ v16bf load_a_frag(const float* rowPtr, int half, int k0) {
  v16bf a;
  const float* p0 = rowPtr + k0 + half * 16;
#pragma unroll
  for (int j = 0; j < 8; ++j) a[j] = (__bf16)p0[j];
  const float* p1 = rowPtr + k0 + half * 16 + 8;
#pragma unroll
  for (int j = 0; j < 8; ++j) a[8 + j] = (__bf16)p1[j];
  return a;
}

// B-fragment (32x16, bf16) when B[k][n] is stored with n-major rows, k contiguous
// (i.e., reading a weight row W[n][k0..k0+31]). lane n = lane&15 (baked into rowPtr),
// element j -> k = half*16 + j
static __device__ __forceinline__ v16bf load_b_frag(const float* rowPtr, int half) {
  v16bf b;
  const float* p = rowPtr + half * 16;
#pragma unroll
  for (int j = 0; j < 16; ++j) b[j] = (__bf16)p[j];
  return b;
}

static __device__ __forceinline__ float max16(float v) {
#pragma unroll
  for (int m = 1; m <= 8; m <<= 1) v = fmaxf(v, __shfl_xor(v, m, 32));
  return v;
}
static __device__ __forceinline__ float sum16(float v) {
#pragma unroll
  for (int m = 1; m <= 8; m <<= 1) v += __shfl_xor(v, m, 32);
  return v;
}
static __device__ __forceinline__ float wave_sum(float v) {
#pragma unroll
  for (int m = 1; m <= 16; m <<= 1) v += __shfl_xor(v, m, 32);
  return v;
}
static __device__ __forceinline__ float wave_max(float v) {
#pragma unroll
  for (int m = 1; m <= 16; m <<= 1) v = fmaxf(v, __shfl_xor(v, m, 32));
  return v;
}

// ---------------------------------------------------------------------------
// Generic GEMM: C[M,N] = A[M,K] @ W[N,K]^T + bias[N]
// Block = 256 threads (8 waves), block tile 128x64, wave tile 16x64.
// W tile staged in LDS as bf16 (coalesced), A fragments straight from global.
// Next K-step prefetched (global_prefetch_b8) while current step computes.
// Requires M%128==0, N%64==0, K%32==0.
// ---------------------------------------------------------------------------
__global__ __launch_bounds__(256) void gemm_kernel(const float* __restrict__ A,
                                                   const float* __restrict__ W,
                                                   const float* __restrict__ bias,
                                                   float* __restrict__ C,
                                                   int M, int N, int K) {
  __shared__ __bf16 sW[64][34];
  const int tid  = threadIdx.x;
  const int lane = tid & 31;
  const int wave = tid >> 5;
  const int half = lane >> 4;
  const int l16  = lane & 15;
  const int n0   = blockIdx.x * 64;
  const int mB   = blockIdx.y * 128 + wave * 16;

  v8f acc[4];
#pragma unroll
  for (int i = 0; i < 4; ++i) acc[i] = v8zero();

  const float* Arow = A + (size_t)(mB + l16) * K;
  const int r  = tid >> 2;        // 0..63  (row of W tile)
  const int kq = (tid & 3) * 8;   // 0,8,16,24

  for (int k0 = 0; k0 < K; k0 += 32) {
    const float* wp = W + (size_t)(n0 + r) * K + k0 + kq;
#pragma unroll
    for (int j = 0; j < 8; ++j) sW[r][kq + j] = (__bf16)wp[j];
    // prefetch next K-step while this one computes
    if (k0 + 32 < K) {
      __builtin_prefetch(wp + 32, 0, 3);
      __builtin_prefetch(Arow + k0 + 32 + half * 8, 0, 3);
    }
    __syncthreads();

    v16bf a = load_a_frag(Arow, half, k0);
#pragma unroll
    for (int nt = 0; nt < 4; ++nt) {
      v16bf b;
      const __bf16* sp = &sW[nt * 16 + l16][half * 16];
#pragma unroll
      for (int j = 0; j < 16; ++j) b[j] = sp[j];
      acc[nt] = wmma_bf16(a, b, acc[nt]);
    }
    __syncthreads();
  }

#pragma unroll
  for (int nt = 0; nt < 4; ++nt) {
    const int n = n0 + nt * 16 + l16;
    const float bv = bias[n];
#pragma unroll
    for (int rr = 0; rr < 8; ++rr) {
      const int m = mB + rr + 8 * half;
      C[(size_t)m * N + n] = acc[nt][rr] + bv;
    }
  }
}

// ---------------------------------------------------------------------------
// Flash attention: one wave per (b, h, 16-query block). Online softmax.
// qkv layout: [row=b*S+s][3*H]: q at h*HD, k at H + h*HD, v at 2H + h*HD.
// ---------------------------------------------------------------------------
__global__ __launch_bounds__(32) void attn_kernel(const float* __restrict__ qkv,
                                                  const float* __restrict__ mask,
                                                  float* __restrict__ ctx) {
  const int bid  = blockIdx.x;              // B*NH*(S/16) = 1024
  const int mt   = bid & 63;
  const int h    = (bid >> 6) & (NHEADS - 1);
  const int b    = bid >> 8;
  const int lane = threadIdx.x & 31;
  const int half = lane >> 4;
  const int l16  = lane & 15;

  __shared__ __bf16 sP[16][34];             // 16 rows x 32 keys prob tile

  const float* qrow = qkv + (size_t)(b * SEQ + mt * 16 + l16) * (3 * H_DIM) + h * HD_DIM;
  v16bf aq[6];
#pragma unroll
  for (int kk = 0; kk < 6; ++kk) aq[kk] = load_a_frag(qrow, half, kk * 32);

  v8f acc[12];
#pragma unroll
  for (int i = 0; i < 12; ++i) acc[i] = v8zero();

  float mrow[8], lrow[8];
#pragma unroll
  for (int rr = 0; rr < 8; ++rr) { mrow[rr] = -1e30f; lrow[rr] = 0.0f; }

  const float scale = 0.07216878364870323f; // 1/sqrt(192)

  for (int step = 0; step < SEQ / 32; ++step) {
    v8f sc[2];
#pragma unroll
    for (int sub = 0; sub < 2; ++sub) {
      sc[sub] = v8zero();
      const int nb = step * 32 + sub * 16;
      const float* krow =
          qkv + (size_t)(b * SEQ + nb + l16) * (3 * H_DIM) + H_DIM + h * HD_DIM;
#pragma unroll
      for (int kk = 0; kk < 6; ++kk) {
        v16bf bk = load_b_frag(krow + kk * 32, half);
        sc[sub] = wmma_bf16(aq[kk], bk, sc[sub]);
      }
      const float pad = mask[b * SEQ + nb + l16];
#pragma unroll
      for (int rr = 0; rr < 8; ++rr) {
        float v = sc[sub][rr] * scale;
        if (pad == 0.0f) v = -1e9f;
        sc[sub][rr] = v;
      }
    }
    // prefetch next step's K rows (this wave reads the same two rows per lane)
    if (step + 1 < SEQ / 32) {
      const float* knext =
          qkv + (size_t)(b * SEQ + (step + 1) * 32 + l16) * (3 * H_DIM) + H_DIM + h * HD_DIM;
      __builtin_prefetch(knext, 0, 3);
      __builtin_prefetch(knext + (size_t)16 * (3 * H_DIM), 0, 3);
    }
    // online softmax update (row = rr + 8*half, replicated across the 16-lane group)
    float alpha[8];
#pragma unroll
    for (int rr = 0; rr < 8; ++rr) {
      float v = fmaxf(sc[0][rr], sc[1][rr]);
      v = max16(v);
      const float nm = fmaxf(mrow[rr], v);
      alpha[rr] = __expf(mrow[rr] - nm);
      const float p0 = __expf(sc[0][rr] - nm);
      const float p1 = __expf(sc[1][rr] - nm);
      sP[rr + 8 * half][l16]      = (__bf16)p0;
      sP[rr + 8 * half][16 + l16] = (__bf16)p1;
      lrow[rr] = lrow[rr] * alpha[rr] + sum16(p0 + p1);
      mrow[rr] = nm;
    }
#pragma unroll
    for (int dt = 0; dt < 12; ++dt)
#pragma unroll
      for (int rr = 0; rr < 8; ++rr) acc[dt][rr] *= alpha[rr];
    __syncthreads();

    // P(16x32) @ V(32x192)
    v16bf ap;
#pragma unroll
    for (int j = 0; j < 16; ++j) {
      const int k = half * 16 + j;
      ap[j] = sP[l16][k];
    }
    const int kb = step * 32;
    // prefetch next step's V rows (strided fragment gathers are the weak point)
    if (step + 1 < SEQ / 32) {
      const float* vnext = qkv + (size_t)(b * SEQ + kb + 32 + lane) * (3 * H_DIM) +
                           2 * H_DIM + h * HD_DIM;
      __builtin_prefetch(vnext, 0, 3);
      __builtin_prefetch(vnext + 128, 0, 3);
    }
#pragma unroll
    for (int dt = 0; dt < 12; ++dt) {
      v16bf bv;
      const float* vp = qkv + (size_t)(b * SEQ + kb + half * 16) * (3 * H_DIM) +
                        2 * H_DIM + h * HD_DIM + dt * 16 + l16;
#pragma unroll
      for (int j = 0; j < 16; ++j) bv[j] = (__bf16)vp[(size_t)j * (3 * H_DIM)];
      acc[dt] = wmma_bf16(ap, bv, acc[dt]);
    }
    __syncthreads();
  }

#pragma unroll
  for (int dt = 0; dt < 12; ++dt) {
#pragma unroll
    for (int rr = 0; rr < 8; ++rr) {
      const int m = mt * 16 + rr + 8 * half;
      ctx[(size_t)(b * SEQ + m) * H_DIM + h * HD_DIM + dt * 16 + l16] =
          acc[dt][rr] / lrow[rr];
    }
  }
}

// ---------------------------------------------------------------------------
// Fused gate + residual + LayerNorm. One block (256 thr) per row of 768.
// Writes enhanced (d_out) and rel_hidden[:, 0:768] (ws).
// ---------------------------------------------------------------------------
__global__ __launch_bounds__(256) void ln_kernel(const float* __restrict__ hs,
                                                 const float* __restrict__ attn,
                                                 const float* __restrict__ gatel,
                                                 const float* __restrict__ g,
                                                 const float* __restrict__ bta,
                                                 float* __restrict__ enh,
                                                 float* __restrict__ relh) {
  const int row = blockIdx.x;
  const int tid = threadIdx.x;
  const size_t base = (size_t)row * H_DIM;
  float x[3];
  float s = 0.0f, s2 = 0.0f;
#pragma unroll
  for (int i = 0; i < 3; ++i) {
    const int c = tid + i * 256;
    const float a  = attn[base + c];
    const float gt = 1.0f / (1.0f + __expf(-gatel[base + c]));
    const float v  = hs[base + c] + gt * a;
    x[i] = v; s += v; s2 += v * v;
  }
  s  = wave_sum(s);
  s2 = wave_sum(s2);
  __shared__ float rs[8], rs2[8];
  const int wv = tid >> 5, ln = tid & 31;
  if (ln == 0) { rs[wv] = s; rs2[wv] = s2; }
  __syncthreads();
  if (wv == 0) {
    float a  = (ln < 8) ? rs[ln]  : 0.0f;
    float b2 = (ln < 8) ? rs2[ln] : 0.0f;
    a = wave_sum(a); b2 = wave_sum(b2);
    if (ln == 0) { rs[0] = a; rs2[0] = b2; }
  }
  __syncthreads();
  const float mean = rs[0] * (1.0f / H_DIM);
  const float var  = rs2[0] * (1.0f / H_DIM) - mean * mean;
  const float inv  = rsqrtf(var + 1e-5f);
#pragma unroll
  for (int i = 0; i < 3; ++i) {
    const int c = tid + i * 256;
    const float y = (x[i] - mean) * inv * g[c] + bta[c];
    enh[base + c] = y;
    relh[(size_t)row * (2 * H_DIM) + c] = y;
  }
}

// ---------------------------------------------------------------------------
// Interleaved RoPE in-place on [row][t*128 + {0..63 q | 64..127 k}].
// One thread per (row, t, pair i).
// ---------------------------------------------------------------------------
__global__ __launch_bounds__(256) void rope_kernel(float* __restrict__ buf, int T) {
  const int idx = blockIdx.x * 256 + threadIdx.x;   // rows*T*32 exactly
  const int i   = idx & 31;
  const int t   = (idx >> 5) % T;
  const int row = idx / (32 * T);
  const int p   = row & (SEQ - 1);
  const float inv = __powf(10000.0f, -(float)i * (1.0f / 32.0f));
  float sn, cs;
  __sincosf((float)p * inv, &sn, &cs);
  float* q = buf + (size_t)row * (T * 2 * D_ROT) + t * (2 * D_ROT) + 2 * i;
  {
    const float x0 = q[0], x1 = q[1];
    q[0] = x0 * cs - x1 * sn;
    q[1] = x1 * cs + x0 * sn;
  }
  float* k = q + D_ROT;
  {
    const float x0 = k[0], x1 = k[1];
    k[0] = x0 * cs - x1 * sn;
    k[1] = x1 * cs + x0 * sn;
  }
}

// ---------------------------------------------------------------------------
// Global-pointer logits: one wave per 16x16 tile of (b,t) batch; K=64 (2 WMMA).
// out[bt][m][n] = (qw[m]·kw[n]*pad[n] - (1-pad[n])*1e12 - (n<m)*1e12) / 8
// ---------------------------------------------------------------------------
__global__ __launch_bounds__(256) void logits_kernel(const float* __restrict__ rot,
                                                     const float* __restrict__ mask,
                                                     float* __restrict__ out, int T) {
  const int wid  = (blockIdx.x * 256 + threadIdx.x) >> 5;
  const int lane = threadIdx.x & 31;
  const int half = lane >> 4;
  const int l16  = lane & 15;
  const int nt    = wid & 63;
  const int mtile = (wid >> 6) & 63;
  const int bt    = wid >> 12;          // b*T + t
  const int b     = bt / T;
  const int t     = bt - b * T;
  const int m0 = mtile * 16, n0 = nt * 16;

  const float* qrow = rot + (size_t)(b * SEQ + m0 + l16) * (T * 2 * D_ROT) + t * (2 * D_ROT);
  const float* krow = rot + (size_t)(b * SEQ + n0 + l16) * (T * 2 * D_ROT) + t * (2 * D_ROT) + D_ROT;

  v8f acc = v8zero();
#pragma unroll
  for (int kk = 0; kk < 64; kk += 32) {
    v16bf a = load_a_frag(qrow, half, kk);
    v16bf bb = load_b_frag(krow + kk, half);
    acc = wmma_bf16(a, bb, acc);
  }

  const int n = n0 + l16;
  const float pad = mask[b * SEQ + n];
  float* op = out + (size_t)bt * SEQ * SEQ;
#pragma unroll
  for (int rr = 0; rr < 8; ++rr) {
    const int m = m0 + rr + 8 * half;
    float v = acc[rr] * pad - (1.0f - pad) * 1e12f;
    if (n < m) v -= 1e12f;
    op[(size_t)m * SEQ + n] = v * 0.125f;   // /sqrt(64)
  }
}

// ---------------------------------------------------------------------------
// ent_prior[(b*S+s)*T + t] = max_n ent_logits[b,t,s,n]. One wave per row.
// ---------------------------------------------------------------------------
__global__ __launch_bounds__(256) void rowmax_kernel(const float* __restrict__ logits,
                                                     float* __restrict__ prior, int T) {
  const int wid  = (blockIdx.x * 256 + threadIdx.x) >> 5;   // B*T*S waves exactly
  const int lane = threadIdx.x & 31;
  const int s  = wid & (SEQ - 1);
  const int t  = (wid >> 10) % T;
  const int b  = wid / (SEQ * T);
  const float* rowp = logits + ((size_t)(b * T + t) * SEQ + s) * SEQ;
  float m = -1e30f;
  for (int c = lane; c < SEQ; c += 32) m = fmaxf(m, rowp[c]);
  m = wave_max(m);
  if (lane == 0) prior[(size_t)(b * SEQ + s) * T + t] = m;
}

// ---------------------------------------------------------------------------
// prior_feat = relu(ent_prior @ prior_w^T + prior_b) -> rel_hidden[:, 768:1536]
// K=9 -> plain VALU (too small for WMMA).
// ---------------------------------------------------------------------------
__global__ __launch_bounds__(256) void prior_kernel(const float* __restrict__ prior,
                                                    const float* __restrict__ pw,
                                                    const float* __restrict__ pb,
                                                    float* __restrict__ relh) {
  const int idx = blockIdx.x * 256 + threadIdx.x;   // ROWS*H exactly
  const int n   = idx % H_DIM;
  const int row = idx / H_DIM;
  float a = pb[n];
#pragma unroll
  for (int e = 0; e < E_T; ++e) a += prior[(size_t)row * E_T + e] * pw[(size_t)n * E_T + e];
  relh[(size_t)row * (2 * H_DIM) + H_DIM + n] = fmaxf(a, 0.0f);
}

// ---------------------------------------------------------------------------
// Host-side orchestration
// ---------------------------------------------------------------------------
extern "C" void kernel_launch(void* const* d_in, const int* in_sizes, int n_in,
                              void* d_out, int out_size, void* d_ws, size_t ws_size,
                              hipStream_t stream) {
  const float* hidden     = (const float*)d_in[0];
  const float* mask       = (const float*)d_in[1];
  const float* in_proj_w  = (const float*)d_in[2];
  const float* in_proj_b  = (const float*)d_in[3];
  const float* out_proj_w = (const float*)d_in[4];
  const float* out_proj_b = (const float*)d_in[5];
  const float* gate_w     = (const float*)d_in[6];
  const float* gate_b     = (const float*)d_in[7];
  const float* ln_g       = (const float*)d_in[8];
  const float* ln_b       = (const float*)d_in[9];
  const float* ent_w      = (const float*)d_in[10];
  const float* ent_b      = (const float*)d_in[11];
  const float* prior_w    = (const float*)d_in[12];
  const float* prior_b    = (const float*)d_in[13];
  const float* rel_w      = (const float*)d_in[14];
  const float* rel_b      = (const float*)d_in[15];

  float* out = (float*)d_out;
  const size_t ENT_OFF = 0;
  const size_t REL_OFF = (size_t)BATCH * E_T * SEQ * SEQ;                 // 37,748,736
  const size_t ENH_OFF = REL_OFF + (size_t)BATCH * R_T * SEQ * SEQ;       // 88,080,384

  float* ws = (float*)d_ws;
  // workspace layout (floats); total ~36.2M floats (~145 MB)
  float* qkv      = ws;                         // 4096*2304
  float* ctx      = qkv      + (size_t)ROWS * 3 * H_DIM;
  float* attn_out = ctx      + (size_t)ROWS * H_DIM;
  float* gate_lin = attn_out + (size_t)ROWS * H_DIM;
  float* rel_hid  = gate_lin + (size_t)ROWS * H_DIM;        // 4096*1536
  float* ent_out  = rel_hid  + (size_t)ROWS * 2 * H_DIM;    // 4096*1152
  float* rel_out  = ent_out  + (size_t)ROWS * (E_T * 2 * D_ROT);  // 4096*1536
  float* ent_pri  = rel_out  + (size_t)ROWS * (R_T * 2 * D_ROT);  // 4096*9

  // 1) qkv = hidden @ in_proj_w^T + b      (4096 x 2304, K=768)
  gemm_kernel<<<dim3(3 * H_DIM / 64, ROWS / 128), 256, 0, stream>>>(
      hidden, in_proj_w, in_proj_b, qkv, ROWS, 3 * H_DIM, H_DIM);

  // 2) flash attention -> ctx (4096 x 768)
  attn_kernel<<<BATCH * NHEADS * (SEQ / 16), 32, 0, stream>>>(qkv, mask, ctx);

  // 3) attn_out = ctx @ out_proj_w^T + b
  gemm_kernel<<<dim3(H_DIM / 64, ROWS / 128), 256, 0, stream>>>(
      ctx, out_proj_w, out_proj_b, attn_out, ROWS, H_DIM, H_DIM);

  // 4) gate_lin = attn_out @ gate_w^T + b
  gemm_kernel<<<dim3(H_DIM / 64, ROWS / 128), 256, 0, stream>>>(
      attn_out, gate_w, gate_b, gate_lin, ROWS, H_DIM, H_DIM);

  // 5) enhanced = LN(hidden + sigmoid(gate_lin)*attn_out)  -> d_out & rel_hidden[:,0:768]
  ln_kernel<<<ROWS, 256, 0, stream>>>(hidden, attn_out, gate_lin, ln_g, ln_b,
                                      out + ENH_OFF, rel_hid);

  // 6) ent_out = enhanced @ ent_w^T + b    (4096 x 1152, K=768)
  gemm_kernel<<<dim3(E_T * 2 * D_ROT / 64, ROWS / 128), 256, 0, stream>>>(
      out + ENH_OFF, ent_w, ent_b, ent_out, ROWS, E_T * 2 * D_ROT, H_DIM);

  // 7) RoPE in place on ent_out
  rope_kernel<<<(ROWS * E_T * 32) / 256, 256, 0, stream>>>(ent_out, E_T);

  // 8) ent_logits -> d_out[ENT_OFF]
  logits_kernel<<<(BATCH * E_T * 64 * 64 * 32) / 256, 256, 0, stream>>>(
      ent_out, mask, out + ENT_OFF, E_T);

  // 9) ent_prior = rowmax(ent_logits)
  rowmax_kernel<<<(BATCH * E_T * SEQ * 32) / 256, 256, 0, stream>>>(
      out + ENT_OFF, ent_pri, E_T);

  // 10) prior_feat = relu(ent_prior @ prior_w^T + b) -> rel_hidden[:,768:]
  prior_kernel<<<(ROWS * H_DIM) / 256, 256, 0, stream>>>(ent_pri, prior_w, prior_b, rel_hid);

  // 11) rel_out = rel_hidden @ rel_w^T + b  (4096 x 1536, K=1536)
  gemm_kernel<<<dim3(R_T * 2 * D_ROT / 64, ROWS / 128), 256, 0, stream>>>(
      rel_hid, rel_w, rel_b, rel_out, ROWS, R_T * 2 * D_ROT, 2 * H_DIM);

  // 12) RoPE in place on rel_out
  rope_kernel<<<(ROWS * R_T * 32) / 256, 256, 0, stream>>>(rel_out, R_T);

  // 13) rel_logits -> d_out[REL_OFF]
  logits_kernel<<<(BATCH * R_T * 64 * 64 * 32) / 256, 256, 0, stream>>>(
      rel_out, mask, out + REL_OFF, R_T);

  (void)in_sizes; (void)n_in; (void)out_size; (void)ws_size;
}
